// SingleHeadAttention_76184129897178
// MI455X (gfx1250) — matrix operand order repaired and checked by reading, not compiled, not run on hardware
//
#include <hip/hip_runtime.h>
#include <hip/hip_bf16.h>
#include <stdint.h>

#define BB 8
#define TT 2048
#define EE 1024
#define AD 1024
#define MM (BB*TT)

typedef __attribute__((ext_vector_type(16))) __bf16 v16bf;
typedef __attribute__((ext_vector_type(8)))  float  v8f;

union Frag {
    v16bf v;
    uint4 u[2];
};

#if defined(__has_builtin)
#if __has_builtin(__builtin_amdgcn_cvt_pk_bf16_f32)
#define HAVE_CVT_PK_BF16 1
#endif
#endif

__device__ __forceinline__ uint32_t pack2bf(float a, float b) {
#ifdef HAVE_CVT_PK_BF16
    auto r = __builtin_amdgcn_cvt_pk_bf16_f32(a, b);
    return __builtin_bit_cast(uint32_t, r);
#else
    uint32_t ua = __float_as_uint(a);
    uint32_t ub = __float_as_uint(b);
    ua = (ua + 0x7FFFu + ((ua >> 16) & 1u)) >> 16;     // RNE truncate to bf16
    ub = (ub + 0x7FFFu + ((ub >> 16) & 1u)) >> 16;
    return (ua & 0xFFFFu) | (ub << 16);
#endif
}
__device__ __forceinline__ uint16_t f2bf(float a) {
#ifdef HAVE_CVT_PK_BF16
    auto r = __builtin_amdgcn_cvt_pk_bf16_f32(a, a);
    return (uint16_t)(__builtin_bit_cast(uint32_t, r) & 0xFFFFu);
#else
    uint32_t ua = __float_as_uint(a);
    return (uint16_t)((ua + 0x7FFFu + ((ua >> 16) & 1u)) >> 16);
#endif
}

// ---------------------------------------------------------------------------
// Kernel 0: bulk fp32 -> bf16 conversion (bandwidth bound)
// ---------------------------------------------------------------------------
__global__ __launch_bounds__(256) void f32_to_bf16_kernel(
    const float* __restrict__ src, uint16_t* __restrict__ dst, int n8)
{
    const int i = blockIdx.x * blockDim.x + threadIdx.x;
    if (i >= n8) return;
    const float4* s = (const float4*)src + (size_t)i * 2;
    const float4 f0 = s[0];
    const float4 f1 = s[1];
    uint4 p;
    p.x = pack2bf(f0.x, f0.y); p.y = pack2bf(f0.z, f0.w);
    p.z = pack2bf(f1.x, f1.y); p.w = pack2bf(f1.z, f1.w);
    ((uint4*)dst)[i] = p;
}

// ---------------------------------------------------------------------------
// Kernel 1 (fast path): QKV projection from pre-converted bf16 operands.
// Pure b128 loads in WMMA fragment layout, no LDS, no packing in the loop.
// grid = (AD/128, MM/128, 3)  z: 0=q(scaled 1/32), 1=k, 2=v(stored transposed)
// ---------------------------------------------------------------------------
__global__ __launch_bounds__(256) void qkv_proj_bf_kernel(
    const uint16_t* __restrict__ Xbf,   // [MM][EE] bf16
    const uint16_t* __restrict__ Wbf,   // [3][AD][EE] bf16
    uint16_t* __restrict__ qbf, uint16_t* __restrict__ kbf,
    uint16_t* __restrict__ vt)
{
    const int z = blockIdx.z;
    const uint16_t* W = Wbf + (size_t)z * AD * EE;
    const float scale = (z == 0) ? 0.03125f : 1.0f;   // 1/sqrt(1024) folded into q

    const int tid   = threadIdx.x;
    const int w     = tid >> 5;
    const int lane  = tid & 31;
    const int lhalf = lane >> 4;
    const int lrow  = lane & 15;
    const int m0 = blockIdx.y * 128 + w * 16;   // this wave's 16 output rows
    const int a0 = blockIdx.x * 128;            // workgroup's 128 output cols

    v8f acc[8];
#pragma unroll
    for (int j = 0; j < 8; ++j)
#pragma unroll
        for (int r = 0; r < 8; ++r) acc[j][r] = 0.0f;

    const uint16_t* xrow = Xbf + (size_t)(m0 + lrow) * EE;

    for (int ke = 0; ke < EE; ke += 32) {
        // A fragment (16x32): two b128 loads in native layout
        Frag af;
        {
            const uint16_t* p = xrow + ke + lhalf * 8;
            af.u[0] = *(const uint4*)p;
            af.u[1] = *(const uint4*)(p + 16);
        }
        // 8 WMMAs along the 128 output columns; B frags direct from global
        // (two 16B loads per lane, 32B apart, L2/WGP$-resident)
#pragma unroll
        for (int j = 0; j < 8; ++j) {
            Frag bfr;
            const uint16_t* p = W + (size_t)(a0 + j * 16 + lrow) * EE + ke + lhalf * 8;
            bfr.u[0] = *(const uint4*)p;
            bfr.u[1] = *(const uint4*)(p + 16);
            acc[j] = __builtin_amdgcn_wmma_f32_16x16x32_bf16(
                false, af.v, false, bfr.v, (short)0, acc[j], false, false);
        }
    }

    if (z == 2) {
        // v stored transposed: vt[b][a][t]; one b128 store per subtile
        const int b = m0 / TT;
        const int t0 = (m0 - b * TT) + lhalf * 8;
#pragma unroll
        for (int j = 0; j < 8; ++j) {
            const int n = a0 + j * 16 + lrow;
            uint4 p;
            p.x = pack2bf(acc[j][0], acc[j][1]);
            p.y = pack2bf(acc[j][2], acc[j][3]);
            p.z = pack2bf(acc[j][4], acc[j][5]);
            p.w = pack2bf(acc[j][6], acc[j][7]);
            *(uint4*)(vt + ((size_t)b * AD + n) * TT + t0) = p;
        }
    } else {
        uint16_t* Y = (z == 0) ? qbf : kbf;
#pragma unroll
        for (int j = 0; j < 8; ++j) {
            const int n = a0 + j * 16 + lrow;
#pragma unroll
            for (int r = 0; r < 8; ++r) {
                const int m = m0 + lhalf * 8 + r;
                Y[(size_t)m * AD + n] = f2bf(acc[j][r] * scale);
            }
        }
    }
}

// ---------------------------------------------------------------------------
// Kernel 1 (fallback, small workspace): convert-in-kernel projection with
// LDS-staged W tile.  Same math, used only if ws can't hold bf16 copies.
// ---------------------------------------------------------------------------
__global__ __launch_bounds__(256) void qkv_proj_kernel(
    const float* __restrict__ X,  const float* __restrict__ Wq,
    const float* __restrict__ Wk, const float* __restrict__ Wv,
    uint16_t* __restrict__ qbf, uint16_t* __restrict__ kbf,
    uint16_t* __restrict__ vt)
{
    __shared__ uint16_t Ws[128][40];      // 128x32 bf16 W tile, rows padded to 80B

    const int z = blockIdx.z;
    const float* W = (z == 0) ? Wq : (z == 1) ? Wk : Wv;
    const float scale = (z == 0) ? 0.03125f : 1.0f;

    const int tid   = threadIdx.x;
    const int w     = tid >> 5;
    const int lane  = tid & 31;
    const int lhalf = lane >> 4;
    const int lrow  = lane & 15;
    const int m0 = blockIdx.y * 128 + w * 16;
    const int a0 = blockIdx.x * 128;

    v8f acc[8];
#pragma unroll
    for (int j = 0; j < 8; ++j)
#pragma unroll
        for (int r = 0; r < 8; ++r) acc[j][r] = 0.0f;

    const int srow  = tid >> 1;
    const int shalf = tid & 1;

    for (int ke = 0; ke < EE; ke += 32) {
        {
            const float* wp = W + (size_t)(a0 + srow) * EE + ke + shalf * 16;
            float4 f0 = ((const float4*)wp)[0];
            float4 f1 = ((const float4*)wp)[1];
            float4 f2 = ((const float4*)wp)[2];
            float4 f3 = ((const float4*)wp)[3];
            uint4 p0, p1;
            p0.x = pack2bf(f0.x, f0.y); p0.y = pack2bf(f0.z, f0.w);
            p0.z = pack2bf(f1.x, f1.y); p0.w = pack2bf(f1.z, f1.w);
            p1.x = pack2bf(f2.x, f2.y); p1.y = pack2bf(f2.z, f2.w);
            p1.z = pack2bf(f3.x, f3.y); p1.w = pack2bf(f3.z, f3.w);
            uint4* dst = (uint4*)&Ws[srow][shalf * 16];
            dst[0] = p0; dst[1] = p1;
        }
        __syncthreads();

        Frag af;
        {
            const float* xa = X + (size_t)(m0 + lrow) * EE + ke + lhalf * 8;
            float4 f0 = ((const float4*)xa)[0];
            float4 f1 = ((const float4*)xa)[1];
            float4 f2 = ((const float4*)(xa + 16))[0];
            float4 f3 = ((const float4*)(xa + 16))[1];
            af.u[0].x = pack2bf(f0.x, f0.y); af.u[0].y = pack2bf(f0.z, f0.w);
            af.u[0].z = pack2bf(f1.x, f1.y); af.u[0].w = pack2bf(f1.z, f1.w);
            af.u[1].x = pack2bf(f2.x, f2.y); af.u[1].y = pack2bf(f2.z, f2.w);
            af.u[1].z = pack2bf(f3.x, f3.y); af.u[1].w = pack2bf(f3.z, f3.w);
        }

#pragma unroll
        for (int j = 0; j < 8; ++j) {
            Frag bfr;
            const uint4* src = (const uint4*)&Ws[j * 16 + lrow][lhalf * 8];
            bfr.u[0] = src[0];
            bfr.u[1] = src[2];
            acc[j] = __builtin_amdgcn_wmma_f32_16x16x32_bf16(
                false, af.v, false, bfr.v, (short)0, acc[j], false, false);
        }
        __syncthreads();
    }

    if (z == 2) {
        const int b = m0 / TT;
        const int t0 = (m0 - b * TT) + lhalf * 8;
#pragma unroll
        for (int j = 0; j < 8; ++j) {
            const int n = a0 + j * 16 + lrow;
            uint4 p;
            p.x = pack2bf(acc[j][0], acc[j][1]);
            p.y = pack2bf(acc[j][2], acc[j][3]);
            p.z = pack2bf(acc[j][4], acc[j][5]);
            p.w = pack2bf(acc[j][6], acc[j][7]);
            *(uint4*)(vt + ((size_t)b * AD + n) * TT + t0) = p;
        }
    } else {
        uint16_t* Y = (z == 0) ? qbf : kbf;
#pragma unroll
        for (int j = 0; j < 8; ++j) {
            const int n = a0 + j * 16 + lrow;
#pragma unroll
            for (int r = 0; r < 8; ++r) {
                const int m = m0 + lhalf * 8 + r;
                Y[(size_t)m * AD + n] = f2bf(acc[j][r] * scale);
            }
        }
    }
}

// ---------------------------------------------------------------------------
// Kernel 2: causal flash attention.  32 query rows / workgroup, 8 waves.
// grid = (TT/32, BB)
// ---------------------------------------------------------------------------
__global__ __launch_bounds__(256) void attn_kernel(
    const uint16_t* __restrict__ qbf, const uint16_t* __restrict__ kbf,
    const uint16_t* __restrict__ vt, float* __restrict__ out)
{
    __shared__ float    Sp[8][16][16];   // per-wave S partials (8 KB)
    __shared__ uint16_t Pl[32][32];      // P in bf16 (2 KB)
    __shared__ float    alpha_s[32];
    __shared__ float    invl_s[32];

    const int b  = blockIdx.y;
    const int q0 = blockIdx.x * 32;
    const int tid   = threadIdx.x;
    const int w     = tid >> 5;
    const int lane  = tid & 31;
    const int lhalf = lane >> 4;
    const int lrow  = lane & 15;

    // S-phase: wave w computes S tile t over E-half h
    const int t   = w & 3;
    const int h   = w >> 2;
    const int qr0 = q0 + (t >> 1) * 16;
    const int kc  = (t & 1) * 16;

    // Q fragments resident in registers (16 x v16bf = 128 VGPRs)
    Frag qf[16];
#pragma unroll
    for (int ks = 0; ks < 16; ++ks) {
        const uint16_t* p = qbf + (size_t)(b * TT + qr0 + lrow) * AD
                            + h * 512 + ks * 32 + lhalf * 8;
        qf[ks].u[0] = *(const uint4*)p;
        qf[ks].u[1] = *(const uint4*)(p + 16);
    }

    // O ownership: rb = row sub-block, cg = 256-col group
    const int rb = w & 1;
    const int cg = w >> 1;
    v8f oacc[16];
#pragma unroll
    for (int j = 0; j < 16; ++j)
#pragma unroll
        for (int r = 0; r < 8; ++r) oacc[j][r] = 0.0f;

    float m_i = -__builtin_inff();
    float l_i = 0.0f;

    const int nkb = q0 / 32 + 1;
    for (int kb = 0; kb < nkb; ++kb) {
        const int k0 = kb * 32;

        // prefetch next key block's K and V lines (global_prefetch_b8)
        if (kb + 1 < nkb) {
            const uint16_t* pk = kbf + (size_t)(b * TT + k0 + 32 + kc + lrow) * AD + h * 512;
            __builtin_prefetch(pk, 0, 3);
            const uint16_t* pv = vt + ((size_t)b * AD + cg * 256 + lane * 8) * TT + k0 + 32;
            __builtin_prefetch(pv, 0, 3);
        }

        // ---- S = Q @ K^T (partial over this wave's E half) ----
        v8f sacc;
#pragma unroll
        for (int r = 0; r < 8; ++r) sacc[r] = 0.0f;
#pragma unroll
        for (int ks = 0; ks < 16; ++ks) {
            Frag kf;
            const uint16_t* p = kbf + (size_t)(b * TT + k0 + kc + lrow) * AD
                                + h * 512 + ks * 32 + lhalf * 8;
            kf.u[0] = *(const uint4*)p;
            kf.u[1] = *(const uint4*)(p + 16);
            sacc = __builtin_amdgcn_wmma_f32_16x16x32_bf16(
                false, qf[ks].v, false, kf.v, (short)0, sacc, false, false);
        }
#pragma unroll
        for (int r = 0; r < 8; ++r)
            Sp[w][lhalf * 8 + r][lrow] = sacc[r];
        __syncthreads();

        // ---- online softmax: wave 0, one lane per query row ----
        if (w == 0) {
            const int r  = lane;
            const int qg = q0 + r;
            float sv[32];
            float mx = m_i;
#pragma unroll
            for (int c = 0; c < 32; ++c) {
                const int tt = ((r >> 4) << 1) | (c >> 4);
                float s = Sp[tt][r & 15][c & 15] + Sp[tt + 4][r & 15][c & 15];
                if (k0 + c > qg) s = -__builtin_inff();   // causal mask
                sv[c] = s;
                mx = fmaxf(mx, s);
            }
            const float al = __expf(m_i - mx);
            float sum = 0.0f;
#pragma unroll
            for (int c = 0; c < 32; ++c) {
                const float e = __expf(sv[c] - mx);
                sum += e;
                Pl[r][c] = f2bf(e);
            }
            l_i = l_i * al + sum;
            m_i = mx;
            alpha_s[r] = al;
        }
        __syncthreads();

        // ---- O = O*alpha + P @ V ----
        float af8[8];
#pragma unroll
        for (int r = 0; r < 8; ++r)
            af8[r] = alpha_s[rb * 16 + lhalf * 8 + r];

        Frag pa;   // P A-fragment for this wave's 16 rows (K = 32 keys)
        {
            const uint16_t* p = &Pl[rb * 16 + lrow][lhalf * 8];
            pa.u[0] = *(const uint4*)p;
            pa.u[1] = *(const uint4*)(p + 16);
        }
#pragma unroll
        for (int j = 0; j < 16; ++j) {
#pragma unroll
            for (int r = 0; r < 8; ++r) oacc[j][r] *= af8[r];
            Frag vb;   // V B-fragment: contiguous keys thanks to transposed vt
            const int col = cg * 256 + j * 16 + lrow;
            const uint16_t* p = vt + ((size_t)b * AD + col) * TT + k0 + lhalf * 8;
            vb.u[0] = *(const uint4*)p;
            vb.u[1] = *(const uint4*)(p + 16);
            oacc[j] = __builtin_amdgcn_wmma_f32_16x16x32_bf16(
                false, pa.v, false, vb.v, (short)0, oacc[j], false, false);
        }
        __syncthreads();   // protect Sp/Pl/alpha for next block
    }

    // ---- normalize and write fp32 output ----
    if (w == 0) invl_s[lane] = 1.0f / l_i;
    __syncthreads();
    float il8[8];
#pragma unroll
    for (int r = 0; r < 8; ++r)
        il8[r] = invl_s[rb * 16 + lhalf * 8 + r];
#pragma unroll
    for (int j = 0; j < 16; ++j) {
        const int col = cg * 256 + j * 16 + lrow;
#pragma unroll
        for (int r = 0; r < 8; ++r) {
            const int row = q0 + rb * 16 + lhalf * 8 + r;
            out[(size_t)(b * TT + row) * AD + col] = oacc[j][r] * il8[r];
        }
    }
}

// ---------------------------------------------------------------------------
extern "C" void kernel_launch(void* const* d_in, const int* in_sizes, int n_in,
                              void* d_out, int out_size, void* d_ws, size_t ws_size,
                              hipStream_t stream) {
    const float* X  = (const float*)d_in[0];   // embedded [B,T,E]
    const float* Wq = (const float*)d_in[1];   // [A,E]
    const float* Wk = (const float*)d_in[2];
    const float* Wv = (const float*)d_in[3];
    float* out = (float*)d_out;                // [B,T,A] fp32

    uint16_t* qbf = (uint16_t*)d_ws;                   // [B,T,A] bf16 (q, pre-scaled)
    uint16_t* kbf = qbf + (size_t)MM * AD;             // [B,T,A] bf16
    uint16_t* vt  = kbf + (size_t)MM * AD;             // [B,A,T] bf16 (transposed)

    const size_t base_elems = 3ull * MM * AD;                       // q,k,vt
    const size_t pre_elems  = base_elems + (size_t)MM * EE + 3ull * AD * EE;

    if (ws_size >= pre_elems * sizeof(uint16_t)) {
        // fast path: bulk-convert operands once, projection is pure bf16 loads
        uint16_t* xbf = vt + (size_t)MM * AD;
        uint16_t* wbf = xbf + (size_t)MM * EE;
        const int nx8 = MM * EE / 8;
        const int nw8 = AD * EE / 8;
        f32_to_bf16_kernel<<<(nx8 + 255) / 256, 256, 0, stream>>>(X, xbf, nx8);
        f32_to_bf16_kernel<<<(nw8 + 255) / 256, 256, 0, stream>>>(Wq, wbf, nw8);
        f32_to_bf16_kernel<<<(nw8 + 255) / 256, 256, 0, stream>>>(Wk, wbf + (size_t)AD * EE, nw8);
        f32_to_bf16_kernel<<<(nw8 + 255) / 256, 256, 0, stream>>>(Wv, wbf + 2ull * AD * EE, nw8);
        qkv_proj_bf_kernel<<<dim3(AD / 128, MM / 128, 3), 256, 0, stream>>>(
            xbf, wbf, qbf, kbf, vt);
    } else {
        // fallback: convert inside the projection kernel
        qkv_proj_kernel<<<dim3(AD / 128, MM / 128, 3), 256, 0, stream>>>(
            X, Wq, Wk, Wv, qbf, kbf, vt);
    }
    attn_kernel<<<dim3(TT / 32, BB), 256, 0, stream>>>(qbf, kbf, vt, out);
}
